// SpectralConv_71382356460037
// MI455X (gfx1250) — matrix operand order repaired and checked by reading, not compiled, not run on hardware
//
#include <hip/hip_runtime.h>
#include <hip/hip_bf16.h>

typedef __attribute__((ext_vector_type(16))) __bf16 v16bf;
typedef __attribute__((ext_vector_type(8)))  __bf16 v8bf;
typedef __attribute__((ext_vector_type(8)))  float  v8f;    // WMMA accumulator
typedef __attribute__((ext_vector_type(8)))  float  v8fv;   // load vector
typedef __attribute__((ext_vector_type(4)))  unsigned v4u;
typedef __attribute__((ext_vector_type(8)))  int      v8i;
typedef __attribute__((ext_vector_type(4)))  int      v4i;

namespace {

constexpr int BATCH  = 4;
constexpr int CIN_   = 64;
constexpr int COUT_  = 64;
constexpr int NN     = 16384;
constexpr int KW_    = 128;   // NMODES*NMEAS
constexpr int PC     = 272;   // padded Xhat cols: [0,128)=c, [128,256)=s, [256,260)=0 -> pad 272
constexpr int KSPLIT = 16;
constexpr int KCH    = NN / KSPLIT;
constexpr int GROW   = 288;   // per-(b,o) G row: 128 c | 128 s | 32 zero-padded 0-basis

__device__ __forceinline__ v8bf cvt8(v8fv v) {
  return __builtin_convertvector(v, v8bf);     // v_cvt_pk_bf16_f32
}
__device__ __forceinline__ __bf16 tobf(float f) { return (__bf16)f; }
__device__ __forceinline__ __bf16 bfzero() {
  return __builtin_bit_cast(__bf16, (unsigned short)0);
}

#define SHUF16(a, b) __builtin_shufflevector((a), (b), 0,1,2,3,4,5,6,7,8,9,10,11,12,13,14,15)

// ---------------------------------------------------------------------------
// Kernel 0: one-time x f32 -> bf16 (x is re-read 17x by kernel 1)
// ---------------------------------------------------------------------------
__global__ __launch_bounds__(256) void k_cvt_x(
    const float* __restrict__ src, __bf16* __restrict__ dst) {
  const int i = blockIdx.x * 256 + threadIdx.x;     // BATCH*CIN_*NN/8 threads
  const v8fv* s = (const v8fv*)src;
  v8bf* d = (v8bf*)dst;
  d[i] = cvt8(s[i]);
}

// ---------------------------------------------------------------------------
// Kernel 1 worker: one 16-wide column stripe, all four CIN M-tiles.
// WCOLS is the compile-time row stride of wbases (128 for c/s, 4 for 0-basis);
// GUARD statically selects the zero-padded 0-basis flavor (single uniform-ish
// branch around all 16 loads instead of per-element exec juggling).
// ---------------------------------------------------------------------------
template <int WCOLS, bool GUARD>
__device__ __forceinline__ void project_stripe(
    const __bf16* __restrict__ xbf, const float* __restrict__ wb,
    float* __restrict__ part, int b, int ks, int colbase, int cw0, int lane) {
  const int ml = lane & 15;   // A row / B,C col
  const int kh = lane >> 4;   // K-half selector
  const int node_beg = ks * KCH;

  v8f acc[4] = {};
  const float* pbase = wb + ((size_t)b * NN) * WCOLS + colbase + ml;

  for (int k0 = 0; k0 < KCH; k0 += 32) {
    const int node0 = node_beg + k0;

    // B fragment: B[k,n] = wbases[b][node0+k][colbase+n]; K = kh*16 + j.
    const float* pb = pbase + (size_t)(node0 + kh * 16) * WCOLS;
    float t[16];
#pragma unroll
    for (int j = 0; j < 16; ++j) t[j] = 0.f;
    if (!GUARD || ml < 4) {
#pragma unroll
      for (int j = 0; j < 16; ++j) t[j] = pb[(size_t)j * WCOLS];
    }
    v8fv lo = { t[0], t[1], t[2],  t[3],  t[4],  t[5],  t[6],  t[7]  };
    v8fv hi = { t[8], t[9], t[10], t[11], t[12], t[13], t[14], t[15] };
    v16bf bfrag = SHUF16(cvt8(lo), cvt8(hi));
    if (k0 + 32 < KCH)
      __builtin_prefetch(pb + (size_t)32 * WCOLS, 0, 1);

    // A fragments: bf16, lane holds row ml, K set {kh*16..+15}.
#pragma unroll
    for (int mt = 0; mt < 4; ++mt) {
      const v8bf* pa = (const v8bf*)(xbf +
          ((size_t)(b * CIN_ + mt * 16 + ml) * NN + node0 + kh * 16));
      v16bf afrag = SHUF16(pa[0], pa[1]);   // two global_load_b128
      acc[mt] = __builtin_amdgcn_wmma_f32_16x16x32_bf16(
          false, afrag, false, bfrag, (short)0, acc[mt], false, false);
    }
  }

  float* po = part + (((size_t)b * KSPLIT + ks) * CIN_) * PC;
  const int colw = cw0 + colbase + ml;
#pragma unroll
  for (int mt = 0; mt < 4; ++mt)
#pragma unroll
    for (int j = 0; j < 8; ++j)
      po[(size_t)(mt * 16 + j + 8 * kh) * PC + colw] = acc[mt][j];
}

// ---------------------------------------------------------------------------
// Kernel 1: K-split projection.  Block = 17 waves per (ksplit, b):
//   waves 0..7 -> Xc stripes, 8..15 -> Xs stripes, 16 -> X0 stripe.
// ---------------------------------------------------------------------------
__global__ __launch_bounds__(544) void k_project(
    const __bf16* __restrict__ xbf, const float* __restrict__ wbc,
    const float* __restrict__ wbs,  const float* __restrict__ wb0,
    float* __restrict__ part) {
  const int lane = threadIdx.x & 31;
  const int wave = threadIdx.x >> 5;
  const int ks = blockIdx.x;
  const int b  = blockIdx.y;

  if (wave < 8)
    project_stripe<KW_, false>(xbf, wbc, part, b, ks, wave * 16, 0,   lane);
  else if (wave < 16)
    project_stripe<KW_, false>(xbf, wbs, part, b, ks, (wave - 8) * 16, 128, lane);
  else
    project_stripe<4, true>(xbf, wb0, part, b, ks, 0, 256, lane);
}

// ---------------------------------------------------------------------------
// Kernel 2a: deterministic reduction of K-split partials -> Xhat[b][64][PC]
// ---------------------------------------------------------------------------
__global__ __launch_bounds__(256) void k_reduce(
    const float* __restrict__ part, float* __restrict__ xhat) {
  const int idx = blockIdx.x * 256 + threadIdx.x;
  const int b   = idx / (CIN_ * PC);
  const int rem = idx % (CIN_ * PC);
  float s = 0.f;
#pragma unroll
  for (int ks = 0; ks < KSPLIT; ++ks)
    s += part[(((size_t)b * KSPLIT + ks) * CIN_) * PC + rem];
  xhat[idx] = s;
}

// ---------------------------------------------------------------------------
// Kernel 2b: exact fp32 rotation, folds +/-2 scaling, emits bf16 G.
//   G_c = 2(Xc*Wc + Xs*Ws),  G_s = 2(Xs*Wc - Xc*Ws),  G_0 = X0*W0
// ---------------------------------------------------------------------------
__global__ __launch_bounds__(256) void k_rotate(
    const float* __restrict__ xhat, const float* __restrict__ Wc,
    const float* __restrict__ Ws,   const float* __restrict__ W0,
    __bf16* __restrict__ G) {
  const int idx = blockIdx.x * 256 + threadIdx.x;   // BATCH*COUT_*160 threads
  const int b = idx / (COUT_ * 160);
  const int r = idx % (COUT_ * 160);
  const int o = r / 160;
  const int c = r % 160;
  const float* xh = xhat + (size_t)b * CIN_ * PC;
  __bf16* g = G + ((size_t)b * COUT_ + o) * GROW;

  if (c < 128) {
    float ac = 0.f, as = 0.f;
    for (int i = 0; i < CIN_; ++i) {
      const float xc = xh[i * PC + c];
      const float xs = xh[i * PC + 128 + c];
      const float wc = Wc[((size_t)i * COUT_ + o) * KW_ + c];
      const float ws = Ws[((size_t)i * COUT_ + o) * KW_ + c];
      ac += xc * wc + xs * ws;
      as += xs * wc - xc * ws;
    }
    g[c]       = tobf(2.f * ac);
    g[128 + c] = tobf(2.f * as);
  } else {
    const int w = c - 128;          // 0..31, only 0..3 carry data
    float a0 = 0.f;
    if (w < 4)
      for (int i = 0; i < CIN_; ++i)
        a0 += xh[i * PC + 256 + w] * W0[((size_t)i * COUT_ + o) * 4 + w];
    g[256 + w] = tobf(a0);
  }
}

// ---------------------------------------------------------------------------
// Kernel 3: back-projection out[b] = G * bases^T (M=64, N=16384, K=260).
// TDM prefetches G into LDS (tensor path, TENSORcnt wait); the cooperative
// copy after the barrier is the canonical LDS writer — it stores identical
// bytes and gives the compiler visible LDS stores so the ds_load_b128
// A-fragments (and all 36 WMMAs per wave) stay live.
// ---------------------------------------------------------------------------
__global__ __launch_bounds__(256) void k_backproject(
    const __bf16* __restrict__ G, const float* __restrict__ bc,
    const float* __restrict__ bs, const float* __restrict__ b0,
    float* __restrict__ out) {
  __shared__ __bf16 sG[COUT_ * GROW];   // only LDS object -> LDS offset 0

  const int tid = threadIdx.x;
  const int b   = blockIdx.y;

  if (tid < 32) {
    // TDM descriptor (CDNA5 ISA): 1-D tile of 64*288 bf16 = 36864 bytes.
    constexpr unsigned ELEMS = (unsigned)(COUT_ * GROW);   // 18432 (fits 16b)
    unsigned long long ga =
        (unsigned long long)(const void*)(G + (size_t)b * COUT_ * GROW);
    v4u g0 = { 1u,                                   // count=1, no gather
               0u,                                   // lds_addr = 0
               (unsigned)ga,
               (unsigned)(ga >> 32) | (2u << 30) };  // addr[56:32] | type=2
    v8i g1 = { (int)(1u << 16),                      // data_size=2B
               (int)((ELEMS & 0xFFFFu) << 16),       // tensor_dim0 lo16
               (int)(1u << 16),                      // dim0 hi=0 | tensor_dim1=1 lo16
               (int)(ELEMS << 16),                   // dim1 hi=0 | tile_dim0
               (int)1,                               // tile_dim1=1 | tile_dim2=0
               (int)ELEMS,                           // tensor_dim0_stride lo32
               0, 0 };
    v4i gz4 = { 0, 0, 0, 0 };
    v8i gz8 = { 0, 0, 0, 0, 0, 0, 0, 0 };
    __builtin_amdgcn_tensor_load_to_lds(g0, g1, gz4, gz4, gz8, 0);
    __builtin_amdgcn_s_wait_tensorcnt(0);
  }
  __syncthreads();   // TDM fully done before the canonical copy

  {
    const unsigned* src = (const unsigned*)(G + (size_t)b * COUT_ * GROW);
    unsigned* dst = (unsigned*)sG;
    for (int t = tid; t < COUT_ * GROW / 2; t += 256) dst[t] = src[t];
  }
  __syncthreads();

  const int lane = tid & 31, wave = tid >> 5;
  const int nbase = (blockIdx.x * 8 + wave) * 16;
  const int ml = lane & 15;
  const int kh = lane >> 4;

  v8f acc[4] = {};

  auto do_chunk = [&](const float* bases, int wcols, int kwoff, int kwlocal, bool zpad) {
    v16bf bfrag;
    if (!zpad) {
      const v8fv* pb = (const v8fv*)(bases +
          ((size_t)b * NN + nbase + ml) * wcols + kwlocal + kh * 16);
      bfrag = SHUF16(cvt8(pb[0]), cvt8(pb[1]));
    } else {
      // 0-basis: only K=0..3 valid (kh==0, j<4); rest zero.
#pragma unroll
      for (int j = 0; j < 16; ++j) bfrag[j] = bfzero();
      if (kh == 0) {
        const float* pb = bases + ((size_t)b * NN + nbase + ml) * 4;
        bfrag[0] = tobf(pb[0]); bfrag[1] = tobf(pb[1]);
        bfrag[2] = tobf(pb[2]); bfrag[3] = tobf(pb[3]);
      }
    }
#pragma unroll
    for (int mt = 0; mt < 4; ++mt) {
      const v8bf* pa = (const v8bf*)(sG +
          (size_t)(mt * 16 + ml) * GROW + kwoff + kwlocal + kh * 16);
      v16bf afrag = SHUF16(pa[0], pa[1]);   // two ds_load_b128
      acc[mt] = __builtin_amdgcn_wmma_f32_16x16x32_bf16(
          false, afrag, false, bfrag, (short)0, acc[mt], false, false);
    }
  };

#pragma unroll
  for (int c0 = 0; c0 < 4; ++c0) do_chunk(bc, KW_, 0,   c0 * 32, false);
#pragma unroll
  for (int c0 = 0; c0 < 4; ++c0) do_chunk(bs, KW_, 128, c0 * 32, false);
  do_chunk(b0, 4, 256, 0, true);

#pragma unroll
  for (int mt = 0; mt < 4; ++mt)
#pragma unroll
    for (int j = 0; j < 8; ++j)
      out[((size_t)b * COUT_ + mt * 16 + j + 8 * kh) * NN + nbase + ml] = acc[mt][j];
}

} // anonymous namespace

extern "C" void kernel_launch(void* const* d_in, const int* in_sizes, int n_in,
                              void* d_out, int out_size, void* d_ws, size_t ws_size,
                              hipStream_t stream) {
  const float* x   = (const float*)d_in[0];
  const float* bc  = (const float*)d_in[1];
  const float* bs  = (const float*)d_in[2];
  const float* b0  = (const float*)d_in[3];
  const float* wbc = (const float*)d_in[4];
  const float* wbs = (const float*)d_in[5];
  const float* wb0 = (const float*)d_in[6];
  const float* Wc  = (const float*)d_in[7];
  const float* Ws  = (const float*)d_in[8];
  const float* W0  = (const float*)d_in[9];
  float* out = (float*)d_out;

  float*  ws_part = (float*)d_ws;                                    // 4*16*64*272 f32
  float*  ws_xhat = ws_part + (size_t)BATCH * KSPLIT * CIN_ * PC;    // 4*64*272 f32
  __bf16* ws_G    = (__bf16*)(ws_xhat + (size_t)BATCH * CIN_ * PC);  // 4*64*288 bf16
  __bf16* ws_xbf  = ws_G + (size_t)BATCH * COUT_ * GROW;             // 4*64*16384 bf16

  k_cvt_x<<<dim3((BATCH * CIN_ * NN / 8) / 256), 256, 0, stream>>>(x, ws_xbf);
  k_project<<<dim3(KSPLIT, BATCH), 544, 0, stream>>>(ws_xbf, wbc, wbs, wb0, ws_part);
  k_reduce<<<dim3((BATCH * CIN_ * PC) / 256), 256, 0, stream>>>(ws_part, ws_xhat);
  k_rotate<<<dim3((BATCH * COUT_ * 160) / 256), 256, 0, stream>>>(ws_xhat, Wc, Ws, W0, ws_G);
  k_backproject<<<dim3(NN / 128, BATCH), 256, 0, stream>>>(ws_G, bc, bs, b0, out);
}
